// GNNEncoder_18064632447517
// MI455X (gfx1250) — compile-verified, matrix operand order checked by reading, not run on hardware
//
#include <hip/hip_runtime.h>
#include <hip/hip_bf16.h>

#define NODES 100000
#define EDGES 1600000
#define DIN   128
#define DHID  32
#define NEG_SLOPE 0.2f

typedef __attribute__((ext_vector_type(16))) __bf16 v16bf;
typedef __attribute__((ext_vector_type(8)))  __bf16 v8bf;
typedef __attribute__((ext_vector_type(8)))  float  v8f;

__device__ __forceinline__ __bf16 f2bf(float f) {
  union { float f; unsigned u; } v; v.f = f;
  // round-to-nearest-even fp32 -> bf16
  unsigned r = (v.u + 0x7FFFu + ((v.u >> 16) & 1u)) >> 16;
  unsigned short hs = (unsigned short)r;
  __bf16 b;
  __builtin_memcpy(&b, &hs, sizeof(b));
  return b;
}

// ---------------------------------------------------------------------------
// Streaming fp32 -> bf16 conversion (row-major, coalesced).
// ---------------------------------------------------------------------------
__global__ __launch_bounds__(256)
void cvt_bf16(const float* __restrict__ in, __bf16* __restrict__ out, long n) {
  const long i = (long)blockIdx.x * blockDim.x + threadIdx.x;
  if (i < n) out[i] = f2bf(in[i]);
}

// ---------------------------------------------------------------------------
// Repack one [K,32] f32 weight into WMMA-B fragment order, bf16:
//   Bprep[((kstep*2 + frag)*32 + lane)*16 + i] = W[k,n]
//   half = lane>>4, n = (lane&15) + 16*frag, k = kstep*32 + half*8 + (i<8?i:i+8)
// Each lane's 16-element fragment becomes one contiguous 32B run.
// ---------------------------------------------------------------------------
__global__ __launch_bounds__(256)
void prep_w(const float* __restrict__ W, __bf16* __restrict__ out, int K) {
  const int idx = blockIdx.x * blockDim.x + threadIdx.x;
  if (idx >= K * DHID) return;
  const int i     = idx & 15;
  const int lane  = (idx >> 4) & 31;
  const int frag  = (idx >> 9) & 1;
  const int kstep = idx >> 10;
  const int half  = lane >> 4;
  const int n     = (lane & 15) + 16 * frag;
  const int kk    = (i < 8) ? i : (i + 8);
  const int k     = kstep * 32 + half * 8 + kk;
  out[idx] = f2bf(W[k * DHID + n]);
}

// ---------------------------------------------------------------------------
// Dual GEMM: Yl = X @ Wl, Yr = X @ Wr. Inputs pre-converted to bf16.
// One wave per 16-row tile; A fragment reused across 4 WMMAs per K-step.
// All fragment loads are wide (b128) vector loads, no conversion VALU.
// ---------------------------------------------------------------------------
template<int K>
__global__ __launch_bounds__(32)
void gemm_dual_wmma(const __bf16* __restrict__ Xbf,
                    const __bf16* __restrict__ Bl,
                    const __bf16* __restrict__ Br,
                    float* __restrict__ Yl,
                    float* __restrict__ Yr) {
  const int  lane = threadIdx.x & 31;
  const int  half = lane >> 4;
  const int  ml   = lane & 15;
  const long tileM = (long)blockIdx.x * 16;

  v8f cl0 = {}, cl1 = {}, cr0 = {}, cr1 = {};

  const __bf16* xrow = Xbf + (tileM + ml) * (long)K + half * 8;
#pragma unroll
  for (int ks = 0; ks < K / 32; ++ks) {
    const v8bf alo = *(const v8bf*)(xrow + ks * 32);       // K 0..7   (+8*half)
    const v8bf ahi = *(const v8bf*)(xrow + ks * 32 + 16);  // K 16..23 (+8*half)
    const v16bf a  = __builtin_shufflevector(alo, ahi,
                      0, 1, 2, 3, 4, 5, 6, 7, 8, 9, 10, 11, 12, 13, 14, 15);
    const long bbase = ((long)(ks * 2) * 32 + lane) * 16;
    const v16bf bl0 = *(const v16bf*)(Bl + bbase);
    const v16bf bl1 = *(const v16bf*)(Bl + bbase + 32 * 16);
    const v16bf br0 = *(const v16bf*)(Br + bbase);
    const v16bf br1 = *(const v16bf*)(Br + bbase + 32 * 16);
    cl0 = __builtin_amdgcn_wmma_f32_16x16x32_bf16(false, a, false, bl0, (short)0, cl0, false, false);
    cl1 = __builtin_amdgcn_wmma_f32_16x16x32_bf16(false, a, false, bl1, (short)0, cl1, false, false);
    cr0 = __builtin_amdgcn_wmma_f32_16x16x32_bf16(false, a, false, br0, (short)0, cr0, false, false);
    cr1 = __builtin_amdgcn_wmma_f32_16x16x32_bf16(false, a, false, br1, (short)0, cr1, false, false);
  }

#pragma unroll
  for (int v = 0; v < 8; ++v) {
    const long row = tileM + v + half * 8;
    Yl[row * DHID + ml]      = cl0[v];
    Yl[row * DHID + ml + 16] = cl1[v];
    Yr[row * DHID + ml]      = cr0[v];
    Yr[row * DHID + ml + 16] = cr1[v];
  }
}

// ---- monotone ordered-uint encoding of float for atomicMax --------------
__device__ __forceinline__ unsigned f2ord(float f) {
  unsigned u = __float_as_uint(f);
  return (u & 0x80000000u) ? ~u : (u | 0x80000000u);
}
__device__ __forceinline__ float ord2f(unsigned o) {
  unsigned u = (o & 0x80000000u) ? (o & 0x7FFFFFFFu) : ~o;
  return __uint_as_float(u);
}

// Per-edge attention logit + segment max. 32 lanes per edge, lane = channel
// -> 128B coalesced gathers; cross-lane reduce via wave32 shfl_xor.
__global__ __launch_bounds__(256)
void edge_logit(const float* __restrict__ xl, const float* __restrict__ xr,
                const float* __restrict__ att,
                const int* __restrict__ src, const int* __restrict__ dst,
                float* __restrict__ e, unsigned* __restrict__ mord) {
  const long tid  = (long)blockIdx.x * blockDim.x + threadIdx.x;
  const long edge = tid >> 5;
  const int  lane = (int)(tid & 31);
  if (edge >= EDGES) return;
  const long s = src[edge];
  const long d = dst[edge];
  float h  = xl[s * DHID + lane] + xr[d * DHID + lane];
  float lr = (h > 0.f) ? h : NEG_SLOPE * h;
  float p  = lr * att[lane];
#pragma unroll
  for (int off = 16; off > 0; off >>= 1) p += __shfl_xor(p, off, 32);
  if (lane == 0) {
    e[edge] = p;
    atomicMax(&mord[d], f2ord(p));
  }
}

// ex = exp(e - m[dst]); segment sum of ex.
__global__ __launch_bounds__(256)
void edge_exp(float* __restrict__ e, const unsigned* __restrict__ mord,
              float* __restrict__ ssum, const int* __restrict__ dst) {
  const long edge = (long)blockIdx.x * blockDim.x + threadIdx.x;
  if (edge >= EDGES) return;
  const long d = dst[edge];
  float m = ord2f(mord[d]);
  if (!__builtin_isfinite(m)) m = 0.f;
  float ex = __expf(e[edge] - m);
  e[edge] = ex;
  atomicAdd(&ssum[d], ex);
}

// out[dst] += alpha * xl[src]; 32 lanes per edge, coalesced 128B atomics.
__global__ __launch_bounds__(256)
void edge_scatter(const float* __restrict__ xl, const float* __restrict__ ex,
                  const float* __restrict__ ssum,
                  const int* __restrict__ src, const int* __restrict__ dst,
                  float* __restrict__ out) {
  const long tid  = (long)blockIdx.x * blockDim.x + threadIdx.x;
  const long edge = tid >> 5;
  const int  lane = (int)(tid & 31);
  if (edge >= EDGES) return;
  const long s = src[edge];
  const long d = dst[edge];
  float alpha = ex[edge] / (ssum[d] + 1e-16f);
  atomicAdd(&out[d * DHID + lane], alpha * xl[s * DHID + lane]);
}

// bias + optional relu; optionally emit bf16 copy (feeds layer-2 GEMM).
__global__ __launch_bounds__(256)
void bias_act(float* __restrict__ y, const float* __restrict__ b, int do_relu,
              __bf16* __restrict__ ybf) {
  const long i = (long)blockIdx.x * blockDim.x + threadIdx.x;
  if (i >= (long)NODES * DHID) return;
  float v = y[i] + b[i & (DHID - 1)];
  if (do_relu) v = fmaxf(v, 0.f);
  y[i] = v;
  if (ybf) ybf[i] = f2bf(v);
}

extern "C" void kernel_launch(void* const* d_in, const int* in_sizes, int n_in,
                              void* d_out, int out_size, void* d_ws, size_t ws_size,
                              hipStream_t stream) {
  const float* x    = (const float*)d_in[0];
  const int*   eidx = (const int*)d_in[1];    // [2, E]: row0=src, row1=dst
  const float* W1l  = (const float*)d_in[2];
  const float* W1r  = (const float*)d_in[3];
  const float* att1 = (const float*)d_in[4];
  const float* b1   = (const float*)d_in[5];
  const float* W2l  = (const float*)d_in[6];
  const float* W2r  = (const float*)d_in[7];
  const float* att2 = (const float*)d_in[8];
  const float* b2   = (const float*)d_in[9];
  const int* src = eidx;
  const int* dst = eidx + EDGES;

  // ---- workspace carve (float region first: 45,600,000 B, 32B-aligned end)
  float*    xl   = (float*)d_ws;                       // N*32
  float*    xr   = xl + (size_t)NODES * DHID;          // N*32
  float*    h    = xr + (size_t)NODES * DHID;          // N*32 (layer-1 out)
  float*    e    = h  + (size_t)NODES * DHID;          // E   (logits -> ex)
  unsigned* mord = (unsigned*)(e + EDGES);             // N   (ordered max)
  float*    ssum = (float*)(mord + NODES);             // N
  __bf16*   xbf  = (__bf16*)(ssum + NODES);            // N*128 bf16 (layer-1 in)
  __bf16*   hbf  = xbf + (size_t)NODES * DIN;          // N*32  bf16 (layer-2 in)
  __bf16*   wlp  = hbf + (size_t)NODES * DHID;         // 128*32 bf16 (B-frag prep)
  __bf16*   wrp  = wlp + DIN * DHID;                   // 128*32 bf16
  float*    out  = (float*)d_out;                      // N*32

  const int MT = NODES / 16;           // 6250 wave tiles
  const int EG = (EDGES * 32) / 256;   // 200000
  const int EB = EDGES / 256;          // 6250
  const int NB = (NODES * DHID) / 256; // 12500

  // ---------------- layer 1 ----------------
  cvt_bf16<<<(NODES * DIN) / 256, 256, 0, stream>>>(x, xbf, (long)NODES * DIN);
  prep_w  <<<(DIN * DHID + 255) / 256, 256, 0, stream>>>(W1l, wlp, DIN);
  prep_w  <<<(DIN * DHID + 255) / 256, 256, 0, stream>>>(W1r, wrp, DIN);
  gemm_dual_wmma<DIN><<<MT, 32, 0, stream>>>(xbf, wlp, wrp, xl, xr);
  hipMemsetAsync(mord, 0, (size_t)NODES * 4, stream);
  hipMemsetAsync(ssum, 0, (size_t)NODES * 4, stream);
  hipMemsetAsync(h,    0, (size_t)NODES * DHID * 4, stream);
  edge_logit  <<<EG, 256, 0, stream>>>(xl, xr, att1, src, dst, e, mord);
  edge_exp    <<<EB, 256, 0, stream>>>(e, mord, ssum, dst);
  edge_scatter<<<EG, 256, 0, stream>>>(xl, e, ssum, src, dst, h);
  bias_act    <<<NB, 256, 0, stream>>>(h, b1, 1, hbf);

  // ---------------- layer 2 ----------------
  prep_w<<<(DHID * DHID + 255) / 256, 256, 0, stream>>>(W2l, wlp, DHID);
  prep_w<<<(DHID * DHID + 255) / 256, 256, 0, stream>>>(W2r, wrp, DHID);
  gemm_dual_wmma<DHID><<<MT, 32, 0, stream>>>(hbf, wlp, wrp, xl, xr);
  hipMemsetAsync(mord, 0, (size_t)NODES * 4, stream);
  hipMemsetAsync(ssum, 0, (size_t)NODES * 4, stream);
  hipMemsetAsync(out,  0, (size_t)NODES * DHID * 4, stream);
  edge_logit  <<<EG, 256, 0, stream>>>(xl, xr, att2, src, dst, e, mord);
  edge_exp    <<<EB, 256, 0, stream>>>(e, mord, ssum, dst);
  edge_scatter<<<EG, 256, 0, stream>>>(xl, e, ssum, src, dst, out);
  bias_act    <<<NB, 256, 0, stream>>>(out, b2, 0, (__bf16*)0);
}
